// MultiHeadAttention_3856880632408
// MI455X (gfx1250) — compile-verified, matrix operand order checked by reading, not run on hardware
//
#include <hip/hip_runtime.h>

typedef __attribute__((ext_vector_type(16))) __bf16 v16bf;
typedef __attribute__((ext_vector_type(8)))  __bf16 v8bf;
typedef __attribute__((ext_vector_type(8)))  float  v8f;
typedef __attribute__((ext_vector_type(4)))  unsigned u32x4;
typedef __attribute__((ext_vector_type(8)))  int      i32x8;
typedef __attribute__((ext_vector_type(4)))  int      i32x4;

#define B_  4
#define S_  2048
#define D_  768
#define H_  12
#define HD_ 64
#define BH_ (B_*H_)
#define M_  (B_*S_)
#define N1_ (3*D_)

__device__ __forceinline__ unsigned short bf16_bits(float f) {
  unsigned u = __builtin_bit_cast(unsigned, f);
  return (unsigned short)((u + 0x7FFFu + ((u >> 16) & 1u)) >> 16);
}
__device__ __forceinline__ __bf16 to_bf16(float f) {
  unsigned short s = bf16_bits(f);
  return __builtin_bit_cast(__bf16, s);
}
__device__ __forceinline__ v8f vzero8() {
  v8f z;
#pragma unroll
  for (int i = 0; i < 8; ++i) z[i] = 0.0f;
  return z;
}

// A-fragment loader for 16x32 bf16 (also used for B via B^T stored row-major).
__device__ __forceinline__ v16bf frag_ld(const __bf16* tile, int stride, int lane) {
  const int r  = lane & 15;
  const int ko = (lane >> 4) << 3;              // 0 or 8
  const __bf16* p = tile + r * stride + ko;
  v8bf lo = *(const v8bf*)(p);                  // K = ko..ko+7
  v8bf hi = *(const v8bf*)(p + 16);             // K = 16+ko..16+ko+7
  v16bf a;
#pragma unroll
  for (int i = 0; i < 8; ++i) { a[i] = lo[i]; a[i + 8] = hi[i]; }
  return a;
}

__device__ __forceinline__ v8f wmma_bf16(v16bf a, v16bf b, v8f c) {
  return __builtin_amdgcn_wmma_f32_16x16x32_bf16(false, a, false, b, (short)0, c,
                                                 false, false);
}

// ---- CDNA5 async global->LDS copy (ASYNCcnt-tracked, no VGPR round trip) ----
__device__ __forceinline__ void async_copy_b128(const void* gptr, void* lptr) {
  unsigned lds = (unsigned)(size_t)lptr;                 // low 32 bits = LDS offset
  unsigned long long g = (unsigned long long)(size_t)gptr;
  asm volatile("global_load_async_to_lds_b128 %0, %1, off"
               :: "v"(lds), "v"(g) : "memory");
}
__device__ __forceinline__ void async_wait0() {
  asm volatile("s_wait_asynccnt 0x0" ::: "memory");
}

// ---- CDNA5 Tensor Data Mover: 2D bf16 tile -> LDS with 16B/row padding ----
// Pads every 32 stored DWORDs (=64 bf16 row) with 4 DWORDs (=8 bf16) -> stride 72.
__device__ __forceinline__ void tdm_load_2d_bf16(const void* gsrc, void* ldst,
                                                 unsigned tensor_d0, unsigned tensor_d1,
                                                 unsigned stride0_elems) {
  unsigned long long ga = (unsigned long long)(size_t)gsrc;
  u32x4 g0;
  g0.x = 1u;                                               // count=1, user mode
  g0.y = (unsigned)(size_t)ldst;                           // LDS byte address
  g0.z = (unsigned)ga;                                     // global_addr[31:0]
  g0.w = (unsigned)((ga >> 32) & 0x01FFFFFFull) | (2u << 30);  // addr[56:32] | type=2
  i32x8 g1;
  g1[0] = (int)((1u << 16) | (1u << 20) | (4u << 22) | (3u << 25));
  //             data_size=2B  pad_en     pad_int=32DW  pad_amt=4DW
  g1[1] = (int)(tensor_d0 << 16);                          // tensor_dim0 lo16
  g1[2] = (int)((tensor_d0 >> 16) | (tensor_d1 << 16));    // dim0 hi16 | dim1 lo16
  g1[3] = (int)((tensor_d1 >> 16) | (64u << 16));          // dim1 hi16 | tile_dim0=64
  g1[4] = (int)64;                                         // tile_dim1=64, tile_dim2=0
  g1[5] = (int)stride0_elems;                              // tensor_dim0_stride lo32
  g1[6] = 0;
  g1[7] = 0;
  i32x4 z4 = {0, 0, 0, 0};
  i32x8 z8 = {0, 0, 0, 0, 0, 0, 0, 0};
  __builtin_amdgcn_tensor_load_to_lds(g0, g1, z4, z4, z8, 0);
}
__device__ __forceinline__ void tdm_wait0() {
  __builtin_amdgcn_s_wait_tensorcnt((short)0);
}

// ---------------- fp32 -> bf16 conversion (x) ----------------
__global__ __launch_bounds__(256) void cvt_f32_bf16(const float* __restrict__ src,
                                                    __bf16* __restrict__ dst, int n4) {
  int i = blockIdx.x * blockDim.x + threadIdx.x;
  if (i >= n4) return;
  const float4 f = ((const float4*)src)[i];
  ushort4 r;
  r.x = bf16_bits(f.x); r.y = bf16_bits(f.y);
  r.z = bf16_bits(f.z); r.w = bf16_bits(f.w);
  ((ushort4*)dst)[i] = r;
}

// ---------------- fp32 [K][N] -> bf16 W^T [N][K] (tiled transpose) ----------------
__global__ __launch_bounds__(256) void cvt_transpose_bf16(const float* __restrict__ W,
                                                          __bf16* __restrict__ Wt,
                                                          int K, int N) {
  __shared__ __bf16 t[32][33];
  const int bn = blockIdx.x * 32, bk = blockIdx.y * 32;
  const int tx = threadIdx.x & 31, ty = threadIdx.x >> 5;   // 32 x 8
#pragma unroll
  for (int r = ty; r < 32; r += 8)
    t[r][tx] = to_bf16(W[(size_t)(bk + r) * N + bn + tx]);  // coalesced read along n
  __syncthreads();
#pragma unroll
  for (int r = ty; r < 32; r += 8)
    Wt[(size_t)(bn + r) * K + bk + tx] = t[tx][r];          // coalesced write along k
}

// ---------------- QKV projection GEMM: [8192,768] x W^T[2304,768] ----------------
// Block tile 128x128, 8 waves x (32x64); async double-buffered LDS staging.
__global__ __launch_bounds__(256) void mha_qkv_gemm(
    const __bf16* __restrict__ X, const __bf16* __restrict__ Wt,
    const float* __restrict__ bias,
    __bf16* __restrict__ Q, __bf16* __restrict__ K, __bf16* __restrict__ VT) {
  __shared__ __align__(16) __bf16 As[2][128][40];
  __shared__ __align__(16) __bf16 Bs[2][128][40];
  const int tid = threadIdx.x;
  const int lane = tid & 31, wave = tid >> 5;
  const int wm = wave & 3, wn = wave >> 2;
  const int mB = blockIdx.x * 128, nB = blockIdx.y * 128;

  v8f acc[2][4];
#pragma unroll
  for (int i = 0; i < 2; ++i)
#pragma unroll
    for (int j = 0; j < 4; ++j) acc[i][j] = vzero8();

  const int srow = tid >> 1, scb = (tid & 1) << 4;
  const __bf16* ag = X  + (size_t)(mB + srow) * D_ + scb;
  const __bf16* bg = Wt + (size_t)(nB + srow) * D_ + scb;

  async_copy_b128(ag,     &As[0][srow][scb]);
  async_copy_b128(ag + 8, &As[0][srow][scb + 8]);
  async_copy_b128(bg,     &Bs[0][srow][scb]);
  async_copy_b128(bg + 8, &Bs[0][srow][scb + 8]);

  for (int kt = 0; kt < D_; kt += 32) {
    const int cur = (kt >> 5) & 1;
    async_wait0();
    __syncthreads();
    if (kt + 32 < D_) {
      const int nxt = cur ^ 1;
      async_copy_b128(ag + kt + 32, &As[nxt][srow][scb]);
      async_copy_b128(ag + kt + 40, &As[nxt][srow][scb + 8]);
      async_copy_b128(bg + kt + 32, &Bs[nxt][srow][scb]);
      async_copy_b128(bg + kt + 40, &Bs[nxt][srow][scb + 8]);
    }
    v16bf a0 = frag_ld(&As[cur][wm * 32][0],      40, lane);
    v16bf a1 = frag_ld(&As[cur][wm * 32 + 16][0], 40, lane);
#pragma unroll
    for (int j = 0; j < 4; ++j) {
      v16bf b = frag_ld(&Bs[cur][wn * 64 + j * 16][0], 40, lane);
      acc[0][j] = wmma_bf16(a0, b, acc[0][j]);
      acc[1][j] = wmma_bf16(a1, b, acc[1][j]);
    }
    __syncthreads();
  }

  const int col = lane & 15, rowAdd = (lane >> 4) << 3;
#pragma unroll
  for (int i = 0; i < 2; ++i)
#pragma unroll
    for (int j = 0; j < 4; ++j) {
      const int n = nB + wn * 64 + j * 16 + col;
      const float bv = bias[n];
      const int part = n / D_;
      const int c = n - part * D_;
      const int h = c >> 6, hd = c & 63;
#pragma unroll
      for (int v = 0; v < 8; ++v) {
        const int mg = mB + wm * 32 + i * 16 + v + rowAdd;
        const int bb = mg >> 11, ss = mg & (S_ - 1);
        const float val = acc[i][j][v] + bv;
        const size_t bh = (size_t)(bb * H_ + h);
        if (part == 0)      Q[(bh * S_ + ss) * HD_ + hd] = to_bf16(val * 0.125f);
        else if (part == 1) K[(bh * S_ + ss) * HD_ + hd] = to_bf16(val);
        else                VT[(bh * HD_ + hd) * S_ + ss] = to_bf16(val);
      }
    }
}

// ---------------- Flash attention ----------------
// 256 threads = 8 waves x 16 q-rows = 128 q-rows/block.
// K/V tiles double-buffered in LDS, filled by TDM (tensor_load_to_lds, TENSORcnt).
__global__ __launch_bounds__(256) void mha_flash(
    const __bf16* __restrict__ Qg, const __bf16* __restrict__ Kg,
    const __bf16* __restrict__ VTg, __bf16* __restrict__ Og) {
  __shared__ __align__(16) __bf16 KT[2][64][72];   // keys x hd
  __shared__ __align__(16) __bf16 VT[2][64][72];   // hd x keys (V^T)
  __shared__ __align__(16) __bf16 P[8][16][72];    // per-wave P tile
  const int tid = threadIdx.x;
  const int lane = tid & 31, wave = tid >> 5;
  const int bh = blockIdx.y;
  const int qbase = blockIdx.x * 128 + wave * 16;
  const int col = lane & 15, rowAdd = (lane >> 4) << 3;

  const __bf16* qtb = Qg + ((size_t)bh * S_ + qbase) * HD_;
  const v16bf qf0 = frag_ld(qtb,      HD_, lane);
  const v16bf qf1 = frag_ld(qtb + 32, HD_, lane);

  const __bf16* kslab = Kg  + (size_t)bh * S_ * HD_;   // [S][64]
  const __bf16* vslab = VTg + (size_t)bh * HD_ * S_;   // [64][S]

  float mstat[8], lstat[8];
  v8f o[4];
#pragma unroll
  for (int v = 0; v < 8; ++v) { mstat[v] = -3.0e30f; lstat[v] = 0.0f; }
#pragma unroll
  for (int t = 0; t < 4; ++t) o[t] = vzero8();

  // prologue: TDM DMA for key-tile 0 into buffer 0 (wave 0 issues)
  if (wave == 0) {
    tdm_load_2d_bf16(kslab, &KT[0][0][0], HD_, S_, HD_);
    tdm_load_2d_bf16(vslab, &VT[0][0][0], S_, HD_, S_);
  }

  for (int kt = 0; kt < S_; kt += 64) {
    const int cur = (kt >> 6) & 1;
    if (wave == 0) tdm_wait0();
    __syncthreads();                      // current buffer ready for everyone

    if (kt + 64 < S_ && wave == 0) {      // overlap next tile's DMA with compute
      const int nxt = cur ^ 1;
      tdm_load_2d_bf16(kslab + (size_t)(kt + 64) * HD_, &KT[nxt][0][0], HD_, S_, HD_);
      tdm_load_2d_bf16(vslab + (kt + 64),               &VT[nxt][0][0], S_, HD_, S_);
    }

    // scores: 16 q rows x 64 keys (Q pre-scaled by 1/sqrt(HD))
    v8f s[4];
#pragma unroll
    for (int sub = 0; sub < 4; ++sub) {
      v8f a = vzero8();
      a = wmma_bf16(qf0, frag_ld(&KT[cur][sub * 16][0],  72, lane), a);
      a = wmma_bf16(qf1, frag_ld(&KT[cur][sub * 16][32], 72, lane), a);
      s[sub] = a;
    }

    // online softmax; row r = v + 8*(lane>=16) lives across 16 lanes
#pragma unroll
    for (int v = 0; v < 8; ++v) {
      float rmax = fmaxf(fmaxf(s[0][v], s[1][v]), fmaxf(s[2][v], s[3][v]));
#pragma unroll
      for (int msk = 1; msk < 16; msk <<= 1)
        rmax = fmaxf(rmax, __shfl_xor(rmax, msk, 16));
      const float mnew = fmaxf(mstat[v], rmax);
      const float alpha = __expf(mstat[v] - mnew);
      mstat[v] = mnew;
      float rsum = 0.0f;
#pragma unroll
      for (int sub = 0; sub < 4; ++sub) {
        const float p = __expf(s[sub][v] - mnew);
        s[sub][v] = p;
        rsum += p;
      }
#pragma unroll
      for (int msk = 1; msk < 16; msk <<= 1)
        rsum += __shfl_xor(rsum, msk, 16);
      lstat[v] = lstat[v] * alpha + rsum;
#pragma unroll
      for (int t = 0; t < 4; ++t) o[t][v] *= alpha;
    }

    // D-layout f32 -> A-layout bf16 via per-wave LDS round trip
#pragma unroll
    for (int sub = 0; sub < 4; ++sub)
#pragma unroll
      for (int v = 0; v < 8; ++v)
        P[wave][v + rowAdd][sub * 16 + col] = to_bf16(s[sub][v]);

    const v16bf p0 = frag_ld(&P[wave][0][0],  72, lane);
    const v16bf p1 = frag_ld(&P[wave][0][32], 72, lane);
#pragma unroll
    for (int hs = 0; hs < 4; ++hs) {
      o[hs] = wmma_bf16(p0, frag_ld(&VT[cur][hs * 16][0],  72, lane), o[hs]);
      o[hs] = wmma_bf16(p1, frag_ld(&VT[cur][hs * 16][32], 72, lane), o[hs]);
    }
    __syncthreads();                      // all reads of cur done
  }

  const int bb = bh / H_, hh = bh - bb * H_;
#pragma unroll
  for (int hs = 0; hs < 4; ++hs)
#pragma unroll
    for (int v = 0; v < 8; ++v) {
      const int qi = qbase + v + rowAdd;
      const int cl = hh * HD_ + hs * 16 + col;
      Og[((size_t)bb * S_ + qi) * D_ + cl] = to_bf16(o[hs][v] / lstat[v]);
    }
}

// ---------------- Output projection GEMM: [8192,768] x W^T[768,768] + bias ----------------
__global__ __launch_bounds__(256) void mha_out_gemm(
    const __bf16* __restrict__ A, const __bf16* __restrict__ Wt,
    const float* __restrict__ bias, float* __restrict__ Out) {
  __shared__ __align__(16) __bf16 As[2][128][40];
  __shared__ __align__(16) __bf16 Bs[2][128][40];
  const int tid = threadIdx.x;
  const int lane = tid & 31, wave = tid >> 5;
  const int wm = wave & 3, wn = wave >> 2;
  const int mB = blockIdx.x * 128, nB = blockIdx.y * 128;

  v8f acc[2][4];
#pragma unroll
  for (int i = 0; i < 2; ++i)
#pragma unroll
    for (int j = 0; j < 4; ++j) acc[i][j] = vzero8();

  const int srow = tid >> 1, scb = (tid & 1) << 4;
  const __bf16* ag = A  + (size_t)(mB + srow) * D_ + scb;
  const __bf16* bg = Wt + (size_t)(nB + srow) * D_ + scb;

  async_copy_b128(ag,     &As[0][srow][scb]);
  async_copy_b128(ag + 8, &As[0][srow][scb + 8]);
  async_copy_b128(bg,     &Bs[0][srow][scb]);
  async_copy_b128(bg + 8, &Bs[0][srow][scb + 8]);

  for (int kt = 0; kt < D_; kt += 32) {
    const int cur = (kt >> 5) & 1;
    async_wait0();
    __syncthreads();
    if (kt + 32 < D_) {
      const int nxt = cur ^ 1;
      async_copy_b128(ag + kt + 32, &As[nxt][srow][scb]);
      async_copy_b128(ag + kt + 40, &As[nxt][srow][scb + 8]);
      async_copy_b128(bg + kt + 32, &Bs[nxt][srow][scb]);
      async_copy_b128(bg + kt + 40, &Bs[nxt][srow][scb + 8]);
    }
    v16bf a0 = frag_ld(&As[cur][wm * 32][0],      40, lane);
    v16bf a1 = frag_ld(&As[cur][wm * 32 + 16][0], 40, lane);
#pragma unroll
    for (int j = 0; j < 4; ++j) {
      v16bf b = frag_ld(&Bs[cur][wn * 64 + j * 16][0], 40, lane);
      acc[0][j] = wmma_bf16(a0, b, acc[0][j]);
      acc[1][j] = wmma_bf16(a1, b, acc[1][j]);
    }
    __syncthreads();
  }

  const int col = lane & 15, rowAdd = (lane >> 4) << 3;
#pragma unroll
  for (int i = 0; i < 2; ++i)
#pragma unroll
    for (int j = 0; j < 4; ++j) {
      const int n = nB + wn * 64 + j * 16 + col;
      const float bv = bias[n];
#pragma unroll
      for (int v = 0; v < 8; ++v) {
        const int mg = mB + wm * 32 + i * 16 + v + rowAdd;
        Out[(size_t)mg * D_ + n] = acc[i][j][v] + bv;
      }
    }
}

extern "C" void kernel_launch(void* const* d_in, const int* in_sizes, int n_in,
                              void* d_out, int out_size, void* d_ws, size_t ws_size,
                              hipStream_t stream) {
  (void)in_sizes; (void)n_in; (void)out_size; (void)ws_size;
  const float* x    = (const float*)d_in[0];
  const float* Wqkv = (const float*)d_in[1];
  const float* bqkv = (const float*)d_in[2];
  const float* Wout = (const float*)d_in[3];
  const float* bout = (const float*)d_in[4];
  float* out = (float*)d_out;

  char* ws = (char*)d_ws;
  size_t off = 0;
  auto take = [&](size_t elems) -> void* {
    void* p = (void*)(ws + off);
    off += (elems * 2 + 255) & ~(size_t)255;
    return p;
  };
  __bf16* xb    = (__bf16*)take((size_t)M_ * D_);
  __bf16* wqkvt = (__bf16*)take((size_t)N1_ * D_);   // W_qkv^T [2304][768]
  __bf16* woutt = (__bf16*)take((size_t)D_ * D_);    // W_out^T [768][768]
  __bf16* qb    = (__bf16*)take((size_t)BH_ * S_ * HD_);
  __bf16* kb    = (__bf16*)take((size_t)BH_ * S_ * HD_);
  __bf16* vtb   = (__bf16*)take((size_t)BH_ * S_ * HD_);
  __bf16* ab    = (__bf16*)take((size_t)M_ * D_);

  cvt_f32_bf16<<<(M_ * D_ / 4 + 255) / 256, 256, 0, stream>>>(x, xb, M_ * D_ / 4);
  cvt_transpose_bf16<<<dim3(N1_ / 32, D_ / 32), 256, 0, stream>>>(Wqkv, wqkvt, D_, N1_);
  cvt_transpose_bf16<<<dim3(D_ / 32, D_ / 32), 256, 0, stream>>>(Wout, woutt, D_, D_);

  mha_qkv_gemm<<<dim3(M_ / 128, N1_ / 128), 256, 0, stream>>>(xb, wqkvt, bqkv, qb, kb, vtb);
  mha_flash<<<dim3(S_ / 128, BH_), 256, 0, stream>>>(qb, kb, vtb, ab);
  mha_out_gemm<<<dim3(M_ / 128, D_ / 128), 256, 0, stream>>>(ab, woutt, bout, out);
}
